// PointFinbaBlock_42915313222099
// MI455X (gfx1250) — matrix operand order repaired and checked by reading, not compiled, not run on hardware
//
#include <hip/hip_runtime.h>
#include <hip/hip_bf16.h>
#include <cstdint>

// ---- problem constants (match reference) ----
#define Bb 16
#define Nn 2048
#define Kk 8
#define Cc 384
#define ROWS (Bb * Nn)            // 32768
#define C2 (2 * Cc)               // 768
#define EPSf 1e-5f
#define TOT_DIFF ((long long)ROWS * Kk * Cc)   // 100663296

typedef __bf16 bf16_t;
typedef __attribute__((ext_vector_type(16))) __bf16 v16bf;
typedef __attribute__((ext_vector_type(8)))  float  v8f;

// ---- helpers ----
__device__ __forceinline__ float wave_sum(float v) {
#pragma unroll
  for (int o = 16; o > 0; o >>= 1) v += __shfl_xor(v, o, 32);
  return v;
}

__device__ __forceinline__ bf16_t f2bf(float f) {
  union { float f; uint32_t u; } a; a.f = f;
  uint32_t r = a.u + 0x7FFFu + ((a.u >> 16) & 1u);   // RNE
  union { uint16_t u; bf16_t b; } o; o.u = (uint16_t)(r >> 16);
  return o.b;
}

// ---- 0: zero the stats accumulators ----
__global__ void k_zero(float* stats) {
  if (threadIdx.x < 2) stats[threadIdx.x] = 0.0f;
}

// ---- 1: layernorm over C per row, one wave32 per row ----
__global__ __launch_bounds__(256) void k_ln1(const float* __restrict__ x,
                                             const float* __restrict__ g,
                                             const float* __restrict__ b,
                                             float* __restrict__ feat) {
  int row  = blockIdx.x * 8 + (threadIdx.x >> 5);
  int lane = threadIdx.x & 31;
  const float* xr = x + (size_t)row * Cc;
  float v[12];
  float s = 0.f, s2 = 0.f;
#pragma unroll
  for (int j = 0; j < 12; ++j) {
    float t = xr[lane + 32 * j];
    v[j] = t; s += t; s2 += t * t;
  }
  s  = wave_sum(s);
  s2 = wave_sum(s2);
  float mu  = s * (1.f / Cc);
  float var = s2 * (1.f / Cc) - mu * mu;
  float rs  = rsqrtf(var + EPSf);
  float* fr = feat + (size_t)row * Cc;
#pragma unroll
  for (int j = 0; j < 12; ++j) {
    int c = lane + 32 * j;
    fr[c] = (v[j] - mu) * rs * g[c] + b[c];
  }
}

// ---- 2: global sum / sumsq of diff = feat[idx] - feat[center] ----
__global__ __launch_bounds__(256) void k_stats(const float* __restrict__ feat,
                                               const int* __restrict__ idx,
                                               float* __restrict__ stats) {
  const long long T = (long long)gridDim.x * 256;
  long long e = (long long)blockIdx.x * 256 + threadIdx.x;
  float s = 0.f, s2 = 0.f;
  for (; e < TOT_DIFF; e += T) {
    int r = (int)(e / Cc);
    int c = (int)(e - (long long)r * Cc);
    int cen = r >> 3;                       // K = 8
    float d = feat[(size_t)idx[r] * Cc + c] - feat[(size_t)cen * Cc + c];
    s += d; s2 += d * d;
  }
  __shared__ float ls[8], ls2[8];
  float ws  = wave_sum(s);
  float ws2 = wave_sum(s2);
  int w = threadIdx.x >> 5, lane = threadIdx.x & 31;
  if (lane == 0) { ls[w] = ws; ls2[w] = ws2; }
  __syncthreads();
  if (threadIdx.x == 0) {
    float a = 0.f, a2 = 0.f;
#pragma unroll
    for (int i = 0; i < 8; ++i) { a += ls[i]; a2 += ls2[i]; }
    atomicAdd(&stats[0], a);
    atomicAdd(&stats[1], a2);
  }
}

// ---- 3: gather / normalize / affine / gaussian weight / max-pool(K) / LN(768) -> bf16 ----
__global__ __launch_bounds__(128) void k_pool_ln2(const float* __restrict__ feat,
                                                  const int* __restrict__ idx,
                                                  const float* __restrict__ dist,
                                                  const float* __restrict__ alpha,
                                                  const float* __restrict__ beta,
                                                  const float* __restrict__ mlg,
                                                  const float* __restrict__ mlb,
                                                  const float* __restrict__ stats,
                                                  bf16_t* __restrict__ hln) {
  int row = blockIdx.x;                 // (b, n)
  int tid = threadIdx.x;

  __shared__ float s_feat[Cc];
  __shared__ float s_pool[C2];
  __shared__ int   s_idx[Kk];
  __shared__ float s_w[Kk];
  __shared__ float s_red[4], s_red2[4];

  const float* fr = feat + (size_t)row * Cc;
#pragma unroll
  for (int t = 0; t < 3; ++t) s_feat[tid + 128 * t] = fr[tid + 128 * t];
  if (tid < Kk) {
    s_idx[tid] = idx[row * Kk + tid];
    float d = dist[row * Kk + tid];
    s_w[tid] = __expf(-0.5f * d * d);
  }
  // global std (ddof=1) from accumulated sum/sumsq
  double S  = (double)stats[0];
  double S2 = (double)stats[1];
  double M  = (double)TOT_DIFF;
  double var = (S2 - S * S / M) / (M - 1.0);
  float invstd = 1.f / ((float)sqrt(var > 0.0 ? var : 0.0) + EPSf);
  __syncthreads();

#pragma unroll
  for (int t = 0; t < 3; ++t) {
    int c   = tid + 128 * t;
    float cen = s_feat[c];
    float a1 = alpha[c],      b1 = beta[c];
    float a2 = alpha[Cc + c], b2 = beta[Cc + c];
    float m1 = -3.4e38f, m2 = -3.4e38f;
#pragma unroll
    for (int k = 0; k < Kk; ++k) {
      float g  = feat[(size_t)s_idx[k] * Cc + c];
      float wk = s_w[k];
      m1 = fmaxf(m1, (a1 * ((g - cen) * invstd) + b1) * wk);
      m2 = fmaxf(m2, (a2 * cen + b2) * wk);
    }
    s_pool[c]      = m1;
    s_pool[Cc + c] = m2;
  }
  __syncthreads();

  // layernorm over 768
  float s = 0.f, s2 = 0.f;
#pragma unroll
  for (int t = 0; t < 6; ++t) {
    float v = s_pool[tid + 128 * t];
    s += v; s2 += v * v;
  }
  float ws  = wave_sum(s);
  float ws2 = wave_sum(s2);
  int w = tid >> 5, lane = tid & 31;
  if (lane == 0) { s_red[w] = ws; s_red2[w] = ws2; }
  __syncthreads();
  float as = s_red[0] + s_red[1] + s_red[2] + s_red[3];
  float as2 = s_red2[0] + s_red2[1] + s_red2[2] + s_red2[3];
  float mu  = as * (1.f / C2);
  float var2 = as2 * (1.f / C2) - mu * mu;
  float rs  = rsqrtf(var2 + EPSf);

  bf16_t* hr = hln + (size_t)row * C2;
#pragma unroll
  for (int t = 0; t < 6; ++t) {
    int j = tid + 128 * t;
    hr[j] = f2bf((s_pool[j] - mu) * rs * mlg[j] + mlb[j]);
  }
}

// ---- 4: transpose W (768x384 f32, row-major) -> Wt[n][k] bf16 ----
__global__ __launch_bounds__(256) void k_wprep(const float* __restrict__ W,
                                               bf16_t* __restrict__ Wt) {
  int e = blockIdx.x * 256 + threadIdx.x;       // e = n*768 + k
  if (e >= Cc * C2) return;
  int n = e / C2;
  int k = e - n * C2;
  Wt[e] = f2bf(W[(size_t)k * Cc + n]);
}

// ---- 5: tiled GEMM (32768x768)@(768x384) via v_wmma_f32_16x16x32_bf16 ----
// Block = 8 waves = 128(M) x 96(N) tile. B panel DMA'd to LDS (async, double
// buffered) with global_load_async_to_lds_b64 / s_wait_asynccnt (ASYNCcnt path).
#define BLK_M 128
#define BLK_N 96
#define NT 6
#define B_ROW_STRIDE 80   // bytes per n-row in LDS (64B data + 16B pad, 16B aligned)
#define PANEL_B (BLK_N * B_ROW_STRIDE)   // 7680 B

// issue this thread's 3 async b64 copies of the 96x32 B panel for K-step kk
__device__ __forceinline__ void stage_b_async(const bf16_t* __restrict__ Bt,
                                              int nbase, int kk,
                                              uint8_t* s_buf, int tid) {
#pragma unroll
  for (int it = 0; it < 3; ++it) {
    int j = tid + it * 256;                 // 768 qwords total
    int n_row = j >> 3, q = j & 7;
    const uint8_t* g =
        (const uint8_t*)(Bt + (size_t)(nbase + n_row) * C2 + kk) + q * 8;
    // low 32 bits of a generic LDS pointer = wave-relative LDS byte offset
    uint32_t lds_off = (uint32_t)(uintptr_t)(s_buf + n_row * B_ROW_STRIDE + q * 8);
    asm volatile("global_load_async_to_lds_b64 %0, %1, off"
                 :
                 : "v"(lds_off), "v"((uint64_t)(uintptr_t)g)
                 : "memory");
  }
}

__global__ __launch_bounds__(256) void k_gemm(const bf16_t* __restrict__ A,   // hln [row][k]
                                              const bf16_t* __restrict__ Bt,  // Wt  [n][k]
                                              const float* __restrict__ bias,
                                              const float* __restrict__ x,
                                              float* __restrict__ out) {
  __shared__ __align__(16) uint8_t s_b[2][PANEL_B];   // double-buffered, 15360 B

  int tid  = threadIdx.x;
  int w    = tid >> 5;              // wave id -> M subtile
  int lane = tid & 31;
  int half = lane >> 4;
  int l15  = lane & 15;

  int mblk  = blockIdx.x & 255;     // 256 M blocks of 128 rows
  int nblk  = blockIdx.x >> 8;      // 4 N blocks of 96 cols
  int nbase = nblk * BLK_N;
  int rowm  = mblk * BLK_M + w * 16;

  const bf16_t* arow = A + (size_t)(rowm + l15) * C2;

  // prologue: stage panel for kk=0 into buffer 0
  stage_b_async(Bt, nbase, 0, s_b[0], tid);
  asm volatile("s_wait_asynccnt 0x0" ::: "memory");
  __syncthreads();

  v8f acc[NT] = {};
  int cur = 0;
  for (int kk = 0; kk < C2; kk += 32) {
    // prefetch next panel into the other buffer (no wait yet)
    if (kk + 32 < C2) stage_b_async(Bt, nbase, kk + 32, s_b[cur ^ 1], tid);

    // A fragment: one contiguous 32B chunk per lane at k = 16*half (matches B)
    union { uint32_t u[8]; v16bf v; } ua;
#pragma unroll
    for (int r = 0; r < 8; ++r) {
      int ka = 16 * half + (r << 1);
      ua.u[r] = *(const uint32_t*)(arow + kk + ka);
    }

#pragma unroll
    for (int t = 0; t < NT; ++t) {
      // B fragment from LDS: lane -> col n=l15 of subtile t, 32B at 32*half
      const uint32_t* bp =
          (const uint32_t*)(s_b[cur] + (t * 16 + l15) * B_ROW_STRIDE + 32 * half);
      union { uint32_t u[8]; v16bf v; } ub;
#pragma unroll
      for (int r = 0; r < 8; ++r) ub.u[r] = bp[r];
      acc[t] = __builtin_amdgcn_wmma_f32_16x16x32_bf16(
          /*neg_a=*/false, ua.v, /*neg_b=*/false, ub.v,
          /*c_mod=*/(short)0, acc[t], /*reuse_a=*/false, /*reuse_b=*/false);
    }

    // my prefetch done + everyone done reading cur -> flip buffers
    asm volatile("s_wait_asynccnt 0x0" ::: "memory");
    __syncthreads();
    cur ^= 1;
  }

  // epilogue: bias + SiLU + residual
#pragma unroll
  for (int t = 0; t < NT; ++t) {
    int col = nbase + t * 16 + l15;
    float bi = bias[col];
#pragma unroll
    for (int v = 0; v < 8; ++v) {
      // C/D layout: lanes 0-15 -> M=v, lanes 16-31 -> M=v+8; N = lane&15
      int m = rowm + v + (half << 3);
      float val = acc[t][v] + bi;
      float h = val / (1.f + __expf(-val));   // SiLU
      size_t o = (size_t)m * Cc + col;
      out[o] = x[o] + h;
    }
  }
}

// ---- host launcher ----
extern "C" void kernel_launch(void* const* d_in, const int* in_sizes, int n_in,
                              void* d_out, int out_size, void* d_ws, size_t ws_size,
                              hipStream_t stream) {
  const float* x      = (const float*)d_in[0];
  const int*   idx    = (const int*)d_in[1];
  const float* dist   = (const float*)d_in[2];
  const float* ln1_g  = (const float*)d_in[3];
  const float* ln1_b  = (const float*)d_in[4];
  const float* alpha  = (const float*)d_in[5];
  const float* beta   = (const float*)d_in[6];
  const float* mlg    = (const float*)d_in[7];
  const float* mlb    = (const float*)d_in[8];
  const float* W      = (const float*)d_in[9];
  const float* bproj  = (const float*)d_in[10];
  float* out = (float*)d_out;

  // workspace layout (bytes)
  char* ws = (char*)d_ws;
  float*  stats = (float*)ws;                            // 2 floats @ 0
  bf16_t* Wt    = (bf16_t*)(ws + 256);                   // 384*768*2 = 589824 B
  float*  feat  = (float*)(ws + 590080);                 // 32768*384*4 = 50331648 B
  bf16_t* hln   = (bf16_t*)(ws + 50921728);              // 32768*768*2 = 50331648 B

  k_zero<<<1, 32, 0, stream>>>(stats);
  k_ln1<<<ROWS / 8, 256, 0, stream>>>(x, ln1_g, ln1_b, feat);
  k_stats<<<4096, 256, 0, stream>>>(feat, idx, stats);
  k_pool_ln2<<<ROWS, 128, 0, stream>>>(feat, idx, dist, alpha, beta, mlg, mlb, stats, hln);
  k_wprep<<<(Cc * C2 + 255) / 256, 256, 0, stream>>>(W, Wt);
  // grid: 256 M blocks x 4 N blocks = 1024 blocks of 256 threads (8 waves)
  k_gemm<<<1024, 256, 0, stream>>>(hln, Wt, bproj, x, out);
}
